// Distangled_NL_82128364634386
// MI455X (gfx1250) — compile-verified, matrix operand order checked by reading, not run on hardware
//
#include <hip/hip_runtime.h>
#include <math.h>

// Problem dims (fixed by reference)
#define B_  8
#define C_  64
#define N_  4096   // H*W
#define HW_ 4096

typedef __attribute__((ext_vector_type(16))) _Float16 v16h;
typedef __attribute__((ext_vector_type(8)))  float    v8f;
typedef __attribute__((ext_vector_type(4)))  int      v4i_;

union FragH { v16h v; _Float16 h[16]; };
union FragF { v8f  v; float    f[8];  };

// 16-bit A/B fragment K-index for element e (0..15) in lane (wave32):
//   VGPR v=e/2: lanes0-15 K=2v..2v+1 (+16 for v>=4), lanes16-31 +8
// Contiguous in groups of 8 elements -> vectorizes to 2x ds_load_b128.
__device__ __forceinline__ int kidx(int e, int lane) {
    return (((e >> 1) & 3) << 1) + (e & 1) + ((e >> 3) << 4) + ((lane >= 16) ? 8 : 0);
}

// CDNA5 async memory->LDS staging (ASYNCcnt) if the toolchain exposes it.
#if defined(__HIP_DEVICE_COMPILE__) && \
    __has_builtin(__builtin_amdgcn_global_load_async_to_lds_b128) && \
    __has_builtin(__builtin_amdgcn_s_wait_asynccnt)
#define ASYNC_COPY 1
#endif

// ---------------------------------------------------------------------------
// K0: spatial mean of x per (b, cin)
__global__ void __launch_bounds__(256) xmean_kernel(const float* __restrict__ x,
                                                    float* __restrict__ xmean) {
    int bc = blockIdx.x; // b*64 + cin, 512 blocks
    const float* p = x + (size_t)bc * HW_;
    float s = 0.f;
    for (int i = threadIdx.x; i < N_; i += 256) s += p[i];
    __shared__ float red[256];
    red[threadIdx.x] = s; __syncthreads();
    for (int st = 128; st > 0; st >>= 1) {
        if (threadIdx.x < st) red[threadIdx.x] += red[threadIdx.x + st];
        __syncthreads();
    }
    if (threadIdx.x == 0) xmean[bc] = red[0] * (1.f / (float)N_);
}

// ---------------------------------------------------------------------------
// K1: q_mean = Wq x_mean + bq ; k_mean = Wk x_mean + bk  (means are linear in x)
__global__ void qkmean_kernel(const float* __restrict__ wq, const float* __restrict__ bq,
                              const float* __restrict__ wk, const float* __restrict__ bk,
                              const float* __restrict__ xmean,
                              float* __restrict__ qmean, float* __restrict__ kmean) {
    int t = blockIdx.x * blockDim.x + threadIdx.x; // 0..511
    if (t >= B_ * C_) return;
    int b = t >> 6, c = t & 63;
    float aq = bq[c], ak = bk[c];
    #pragma unroll 8
    for (int ci = 0; ci < 64; ++ci) {
        float xm = xmean[b * 64 + ci];
        aq = fmaf(wq[c * 64 + ci], xm, aq);
        ak = fmaf(wk[c * 64 + ci], xm, ak);
    }
    qmean[t] = aq; kmean[t] = ak;
}

// ---------------------------------------------------------------------------
// K2: per-pixel QKV GEMM + whitening + f16 pack + unary logits
//   qh, kh laid out [b][n][64]   (A / K^T-B fragment friendly)
//   vh     laid out [b][d][n]    (transposed: PV B-fragments vectorize in LDS)
__global__ void __launch_bounds__(256) qkv_kernel(
        const float* __restrict__ x,
        const float* __restrict__ wq, const float* __restrict__ bq,
        const float* __restrict__ wk, const float* __restrict__ bk,
        const float* __restrict__ wv, const float* __restrict__ bv,
        const float* __restrict__ qmean, const float* __restrict__ kmean,
        _Float16* __restrict__ qh, _Float16* __restrict__ kh, _Float16* __restrict__ vh,
        float* __restrict__ u) {
    __shared__ float sW[3][64 * 64];
    __shared__ float sB[3][64];
    __shared__ float sQm[64], sKm[64];
    int gid = blockIdx.x * 256 + threadIdx.x;   // 128 blocks x 256
    int b = gid >> 12, n = gid & 4095;          // whole block in one batch (4096%256==0)
    for (int i = threadIdx.x; i < 4096; i += 256) {
        sW[0][i] = wq[i]; sW[1][i] = wk[i]; sW[2][i] = wv[i];
    }
    if (threadIdx.x < 64) {
        sB[0][threadIdx.x] = bq[threadIdx.x];
        sB[1][threadIdx.x] = bk[threadIdx.x];
        sB[2][threadIdx.x] = bv[threadIdx.x];
        sQm[threadIdx.x] = qmean[b * 64 + threadIdx.x];
        sKm[threadIdx.x] = kmean[b * 64 + threadIdx.x];
    }
    __syncthreads();

    float xr[64];
    #pragma unroll
    for (int ci = 0; ci < 64; ++ci) xr[ci] = x[((size_t)(b * 64 + ci)) * HW_ + n];

    float uacc = 0.f;
    size_t outb = ((size_t)b * N_ + n) * 64;
    for (int c = 0; c < 64; ++c) {
        float aq = sB[0][c], ak = sB[1][c], av = sB[2][c];
        #pragma unroll
        for (int ci = 0; ci < 64; ++ci) {
            float xv = xr[ci];
            aq = fmaf(sW[0][c * 64 + ci], xv, aq);
            ak = fmaf(sW[1][c * 64 + ci], xv, ak);
            av = fmaf(sW[2][c * 64 + ci], xv, av);
        }
        float qw = aq - sQm[c];
        float kw = ak - sKm[c];
        qh[outb + c] = (_Float16)(qw * 0.125f);              // fold 1/sqrt(C) into q
        kh[outb + c] = (_Float16)kw;
        vh[((size_t)(b * 64 + c)) * HW_ + n] = (_Float16)av; // transposed V
        uacc = fmaf(sQm[c], kw, uacc);                       // unary logit (unscaled)
    }
    u[b * N_ + n] = uacc;
}

// ---------------------------------------------------------------------------
// K3: unary softmax over keys + out_unary[b,c] = sum_m v[c,m] * p[m]
__global__ void __launch_bounds__(256) unary_kernel(const float* __restrict__ u,
                                                    const _Float16* __restrict__ vh,
                                                    float* __restrict__ uo) {
    int b = blockIdx.x; // 8 blocks
    const float* ub = u + b * N_;
    __shared__ float red[256];
    float m = -INFINITY;
    for (int i = threadIdx.x; i < N_; i += 256) m = fmaxf(m, ub[i]);
    red[threadIdx.x] = m; __syncthreads();
    for (int st = 128; st > 0; st >>= 1) {
        if (threadIdx.x < st) red[threadIdx.x] = fmaxf(red[threadIdx.x], red[threadIdx.x + st]);
        __syncthreads();
    }
    float umax = red[0]; __syncthreads();
    float s = 0.f;
    for (int i = threadIdx.x; i < N_; i += 256) s += __expf(ub[i] - umax);
    red[threadIdx.x] = s; __syncthreads();
    for (int st = 128; st > 0; st >>= 1) {
        if (threadIdx.x < st) red[threadIdx.x] += red[threadIdx.x + st];
        __syncthreads();
    }
    float inv = 1.f / red[0]; __syncthreads();

    int c = threadIdx.x & 63, grp = threadIdx.x >> 6; // 64 channels x 4 key groups
    float acc = 0.f;
    for (int mm = grp * 1024; mm < (grp + 1) * 1024; ++mm) {
        float p = __expf(ub[mm] - umax);
        acc = fmaf(p, (float)vh[((size_t)(b * 64 + c)) * HW_ + mm], acc); // [b][c][m]
    }
    red[threadIdx.x] = acc; __syncthreads();
    if (grp == 0)
        uo[b * 64 + c] = (red[c] + red[64 + c] + red[128 + c] + red[192 + c]) * inv;
}

// ---------------------------------------------------------------------------
// K4: flash attention. 4 waves/block; wave owns 16 query rows x full d=64.
// Double-buffered K/V tiles, async memory->LDS when available.
__global__ void __launch_bounds__(128) attn_kernel(
        const _Float16* __restrict__ qh, const _Float16* __restrict__ kh,
        const _Float16* __restrict__ vh, const float* __restrict__ x,
        const float* __restrict__ uo, float* __restrict__ out) {
    __shared__ _Float16 lK[2][64 * 64];     // key tile  [key][d]
    __shared__ _Float16 lVt[2][64 * 64];    // val tile  [d][key]  (transposed)
    __shared__ _Float16 lP[4][16 * 64];     // per-wave P relay (C-layout -> A-layout)

    int b     = blockIdx.x >> 6;
    int qbase = (blockIdx.x & 63) << 6;
    int wave  = threadIdx.x >> 5;
    int lane  = threadIdx.x & 31;
    int hl    = lane & 15;
    int hgrp  = lane >> 4;
    int qrow  = qbase + wave * 16 + hl;     // A-matrix row for this lane

    // stage one 64-key tile (8 x b128 per thread) into buffer `buf`
    auto stage = [&](int kt, int buf) {
        int kbase = kt << 6;
        const _Float16* gK = kh + ((size_t)b * N_ + kbase) * 64; // 4096 halfs, flat
        #pragma unroll
        for (int i = 0; i < 4; ++i) {
            int idx  = threadIdx.x + i * 128;     // 0..511 chunks of 8 halfs
            int d    = idx >> 3, part = idx & 7;  // V chunk -> (d row, key part)
            const _Float16* gk = gK + idx * 8;
            const _Float16* gv = vh + ((size_t)(b * 64 + d)) * HW_ + kbase + part * 8;
#ifdef ASYNC_COPY
            __builtin_amdgcn_global_load_async_to_lds_b128(
                (__attribute__((address_space(1))) v4i_*)gk,
                (__attribute__((address_space(3))) v4i_*)&lK[buf][idx * 8], 0, 0);
            __builtin_amdgcn_global_load_async_to_lds_b128(
                (__attribute__((address_space(1))) v4i_*)gv,
                (__attribute__((address_space(3))) v4i_*)&lVt[buf][idx * 8], 0, 0);
#else
            *(uint4*)&lK[buf][idx * 8]  = *(const uint4*)gk;
            *(uint4*)&lVt[buf][idx * 8] = *(const uint4*)gv;
#endif
        }
    };

    // Q fragments (A, 16x32 f16 x 2 K-chunks), loaded once
    FragH aq[2];
    #pragma unroll
    for (int kc = 0; kc < 2; ++kc)
        #pragma unroll
        for (int e = 0; e < 16; ++e)
            aq[kc].h[e] = qh[((size_t)b * N_ + qrow) * 64 + kc * 32 + kidx(e, lane)];

    FragF acc[4];
    #pragma unroll
    for (int nb = 0; nb < 4; ++nb) acc[nb].v = (v8f){};
    float mi[8], li[8];
    #pragma unroll
    for (int r = 0; r < 8; ++r) { mi[r] = -INFINITY; li[r] = 0.f; }

    stage(0, 0);

    for (int kt = 0; kt < 64; ++kt) {
        int cur = kt & 1;
#ifdef ASYNC_COPY
        if (kt + 1 < 64) {
            stage(kt + 1, cur ^ 1);                 // 16 in flight
            __builtin_amdgcn_s_wait_asynccnt(8);    // tile kt landed (in-order)
        } else {
            __builtin_amdgcn_s_wait_asynccnt(0);
        }
#else
        if (kt + 1 < 64) stage(kt + 1, cur ^ 1);    // plain loads overlap via LOADcnt
#endif
        __syncthreads();

        // S = Q * K^T  (scale pre-folded into q)
        FragF s[4];
        #pragma unroll
        for (int nb = 0; nb < 4; ++nb) {
            v8f c = (v8f){};
            #pragma unroll
            for (int kc = 0; kc < 2; ++kc) {
                FragH bk;
                #pragma unroll
                for (int e = 0; e < 16; ++e)
                    bk.h[e] = lK[cur][(nb * 16 + hl) * 64 + kc * 32 + kidx(e, lane)];
                c = __builtin_amdgcn_wmma_f32_16x16x32_f16(false, aq[kc].v, false, bk.v,
                                                           (short)0, c, false, false);
            }
            s[nb].v = c;
        }

        // online softmax: row stats per 16-lane half (row = r + 8*hgrp)
        float corr[8];
        #pragma unroll
        for (int r = 0; r < 8; ++r) {
            float tm = fmaxf(fmaxf(s[0].f[r], s[1].f[r]), fmaxf(s[2].f[r], s[3].f[r]));
            #pragma unroll
            for (int msk = 1; msk < 16; msk <<= 1) tm = fmaxf(tm, __shfl_xor(tm, msk, 32));
            float nm = fmaxf(mi[r], tm);
            corr[r] = __expf(mi[r] - nm);
            mi[r] = nm;
        }
        #pragma unroll
        for (int r = 0; r < 8; ++r) {
            float rs = 0.f;
            #pragma unroll
            for (int nb = 0; nb < 4; ++nb) {
                float p = __expf(s[nb].f[r] - mi[r]);
                s[nb].f[r] = p;
                rs += p;
            }
            #pragma unroll
            for (int msk = 1; msk < 16; msk <<= 1) rs += __shfl_xor(rs, msk, 32);
            li[r] = li[r] * corr[r] + rs;
            #pragma unroll
            for (int nb = 0; nb < 4; ++nb) acc[nb].f[r] *= corr[r];
        }

        // relay P through LDS: C layout -> row-major [qrow][key]
        #pragma unroll
        for (int nb = 0; nb < 4; ++nb)
            #pragma unroll
            for (int r = 0; r < 8; ++r)
                lP[wave][(r + 8 * hgrp) * 64 + nb * 16 + hl] = (_Float16)s[nb].f[r];
        __syncthreads();

        // O += P * V   (V in LDS as [d][key] so B fragments vectorize)
        #pragma unroll
        for (int kc = 0; kc < 2; ++kc) {
            FragH ap;
            #pragma unroll
            for (int e = 0; e < 16; ++e)
                ap.h[e] = lP[wave][hl * 64 + kc * 32 + kidx(e, lane)];
            #pragma unroll
            for (int nb = 0; nb < 4; ++nb) {
                FragH bv;
                #pragma unroll
                for (int e = 0; e < 16; ++e)
                    bv.h[e] = lVt[cur][(nb * 16 + hl) * 64 + kc * 32 + kidx(e, lane)];
                acc[nb].v = __builtin_amdgcn_wmma_f32_16x16x32_f16(false, ap.v, false, bv.v,
                                                                   (short)0, acc[nb].v, false, false);
            }
        }
        __syncthreads();   // protect buffer cur before it is restaged at kt+2
    }

    // epilogue: out = x + acc/l + out_unary (row/lane stats already aligned)
    #pragma unroll
    for (int nb = 0; nb < 4; ++nb)
        #pragma unroll
        for (int r = 0; r < 8; ++r) {
            int row = r + 8 * hgrp;
            int c   = nb * 16 + hl;
            int n   = qbase + wave * 16 + row;
            size_t idx = ((size_t)(b * 64 + c)) * HW_ + n;
            out[idx] = x[idx] + acc[nb].f[r] / li[r] + uo[b * 64 + c];
        }
}

// ---------------------------------------------------------------------------
extern "C" void kernel_launch(void* const* d_in, const int* in_sizes, int n_in,
                              void* d_out, int out_size, void* d_ws, size_t ws_size,
                              hipStream_t stream) {
    const float* x  = (const float*)d_in[0];
    const float* wq = (const float*)d_in[1];
    const float* bq = (const float*)d_in[2];
    const float* wk = (const float*)d_in[3];
    const float* bk = (const float*)d_in[4];
    const float* wv = (const float*)d_in[5];
    const float* bv = (const float*)d_in[6];
    float* out = (float*)d_out;

    char* ws = (char*)d_ws;
    size_t off = 0;
    auto alloc = [&](size_t bytes) -> void* {
        void* p = ws + off;
        off += (bytes + 255) & ~(size_t)255;
        return p;
    };
    _Float16* qh   = (_Float16*)alloc((size_t)B_ * N_ * 64 * sizeof(_Float16)); // 4 MB
    _Float16* kh   = (_Float16*)alloc((size_t)B_ * N_ * 64 * sizeof(_Float16)); // 4 MB
    _Float16* vh   = (_Float16*)alloc((size_t)B_ * N_ * 64 * sizeof(_Float16)); // 4 MB
    float* xmean   = (float*)alloc((size_t)B_ * C_ * sizeof(float));
    float* qmean   = (float*)alloc((size_t)B_ * C_ * sizeof(float));
    float* kmean   = (float*)alloc((size_t)B_ * C_ * sizeof(float));
    float* u       = (float*)alloc((size_t)B_ * N_ * sizeof(float));            // 128 KB
    float* uo      = (float*)alloc((size_t)B_ * C_ * sizeof(float));
    (void)ws_size; (void)in_sizes; (void)n_in; (void)out_size;

    xmean_kernel<<<B_ * C_, 256, 0, stream>>>(x, xmean);
    qkmean_kernel<<<2, 256, 0, stream>>>(wq, bq, wk, bk, xmean, qmean, kmean);
    qkv_kernel<<<(B_ * N_) / 256, 256, 0, stream>>>(x, wq, bq, wk, bk, wv, bv,
                                                    qmean, kmean, qh, kh, vh, u);
    unary_kernel<<<B_, 256, 0, stream>>>(u, vh, uo);
    attn_kernel<<<B_ * (N_ / 64), 128, 0, stream>>>(qh, kh, vh, x, uo, out);
}
